// DiagTripleRCell_29497835388994
// MI455X (gfx1250) — matrix operand order repaired and checked by reading, not compiled, not run on hardware
//
#include <hip/hip_runtime.h>

typedef __attribute__((ext_vector_type(2)))  float   v2f;
typedef __attribute__((ext_vector_type(8)))  float   v8f;
typedef __attribute__((ext_vector_type(16))) __bf16  v16bf;
typedef __attribute__((ext_vector_type(8)))  __bf16  v8bf;

#define T_DIM 2048
#define B_DIM 16
#define D_DIM 1024
#define M_DIM (T_DIM * B_DIM)   // 32768

#define TM 128
#define TN 128

// ---------------------------------------------------------------------------
// WMMA wrappers
// ---------------------------------------------------------------------------
__device__ __forceinline__ v8f wmma4(v2f a, v2f b, v8f c) {
  return __builtin_amdgcn_wmma_f32_16x16x4_f32(false, a, false, b, (short)0, c,
                                               false, false);
}
__device__ __forceinline__ v8f wmma32(v16bf a, v16bf b, v8f c) {
  return __builtin_amdgcn_wmma_f32_16x16x32_bf16(false, a, false, b, (short)0,
                                                 c, false, false);
}

// CDNA5 async copy global->LDS (ASYNCcnt path, VGLOBAL encoding:
// VDST = LDS byte address, VADDR = 32-bit byte offset, SADDR = 64-bit base).
__device__ __forceinline__ void async_b128(unsigned lds_addr, const void* base,
                                           unsigned byte_off) {
  asm volatile("global_load_async_to_lds_b128 %0, %1, %2 offset:0"
               :: "v"(lds_addr), "v"(byte_off), "s"(base) : "memory");
}

// ---------------------------------------------------------------------------
// fp32 -> (bf16 hi, bf16 lo) split, RNE both terms.
// ---------------------------------------------------------------------------
__device__ __forceinline__ void split1(float x, unsigned short& h,
                                       unsigned short& l) {
  unsigned int u = __float_as_uint(x);
  unsigned int r = u + 0x7FFFu + ((u >> 16) & 1u);
  h = (unsigned short)(r >> 16);
  float hv = __uint_as_float(r & 0xFFFF0000u);
  float res = x - hv;
  unsigned int v = __float_as_uint(res);
  l = (unsigned short)((v + 0x7FFFu + ((v >> 16) & 1u)) >> 16);
}

__global__ __launch_bounds__(256)
void split_kernel(const float* __restrict__ src, unsigned short* __restrict__ hi,
                  unsigned short* __restrict__ lo, int n4) {
  const int i = blockIdx.x * 256 + threadIdx.x;
  if (i >= n4) return;
  const float4 v = ((const float4*)src)[i];
  ushort4 h, l;
  split1(v.x, h.x, l.x);
  split1(v.y, h.y, l.y);
  split1(v.z, h.z, l.z);
  split1(v.w, h.w, l.w);
  ((ushort4*)hi)[i] = h;
  ((ushort4*)lo)[i] = l;
}

// ---------------------------------------------------------------------------
// bf16x3 GEMM: 4-deep LDS ring fed by global_load_async_to_lds_b128 with two
// K-tiles in flight (async loads complete in-order per wave, so waiting
// asynccnt<=8 guarantees the older tile has landed).
// C[m,n] = sum_k X[m,k]*W[n,k] + bias[n]  (act==1 -> SiLU)
// ---------------------------------------------------------------------------
#define KT 32
#define BSTR (KT + 8)              // 40 halves = 80B row stride
#define BUFB (TM * BSTR * 2)       // 10240 bytes per LDS buffer slice
#define NBUF 4

__device__ __forceinline__ v16bf frag2(const unsigned short* p0,
                                       const unsigned short* p1) {
  union { v16bf v; v8bf h[2]; } u;
  u.h[0] = *(const v8bf*)p0;
  u.h[1] = *(const v8bf*)p1;
  return u.v;
}

__global__ __launch_bounds__(256, 1)
void proj_gemm_bf3(const unsigned short* __restrict__ Xh,
                   const unsigned short* __restrict__ Xl,
                   const unsigned short* __restrict__ Wh,
                   const unsigned short* __restrict__ Wl,
                   const float* __restrict__ bias, float* __restrict__ out,
                   int act) {
  __shared__ __align__(16) unsigned short Ah[NBUF][TM][BSTR];
  __shared__ __align__(16) unsigned short Al[NBUF][TM][BSTR];
  __shared__ __align__(16) unsigned short Bh[NBUF][TN][BSTR];
  __shared__ __align__(16) unsigned short Bl[NBUF][TN][BSTR];   // 160 KB

  const int K = D_DIM, N = D_DIM;
  const int tid = threadIdx.x;
  const int m0 = blockIdx.y * TM;
  const int n0 = blockIdx.x * TN;

  const int lr = tid >> 2;          // 0..63
  const int lc = (tid & 3) << 3;    // 0,8,16,24 (halves)

  const int wave = tid >> 5;
  const int lane = tid & 31;
  const int wm = wave >> 1;         // 0..3 -> 32-row band
  const int wn = wave & 1;          // 0..1 -> 64-col band
  const int l15 = lane & 15;
  const int lhi = lane >> 4;

  v8f acc[2][4] = {};

  // per-thread LDS byte addresses (buffer 0) and global byte offsets
  unsigned ldsAh[2], ldsAl[2], ldsBh[2], ldsBl[2], offA[2], offB[2];
#pragma unroll
  for (int p = 0; p < 2; ++p) {
    const int row = lr + 64 * p;
    ldsAh[p] = (unsigned)(size_t)&Ah[0][row][lc];
    ldsAl[p] = (unsigned)(size_t)&Al[0][row][lc];
    ldsBh[p] = (unsigned)(size_t)&Bh[0][row][lc];
    ldsBl[p] = (unsigned)(size_t)&Bl[0][row][lc];
    offA[p] = (unsigned)(((size_t)(m0 + row) * K + lc) * 2);
    offB[p] = (unsigned)(((size_t)(n0 + row) * K + lc) * 2);
  }

  auto a_issue = [&](int k0, int buf) {
    const unsigned ab = (unsigned)buf * (unsigned)BUFB;
    const unsigned kb = (unsigned)k0 * 2u;
#pragma unroll
    for (int p = 0; p < 2; ++p) {
      async_b128(ldsAh[p] + ab, Xh, offA[p] + kb);
      async_b128(ldsAl[p] + ab, Xl, offA[p] + kb);
      async_b128(ldsBh[p] + ab, Wh, offB[p] + kb);
      async_b128(ldsBl[p] + ab, Wl, offB[p] + kb);
    }
  };

  const int NTILES = K / KT;
  a_issue(0, 0);
  a_issue(KT, 1);
  for (int t = 0; t < NTILES; ++t) {
    const int cur = t & (NBUF - 1);
    // Tile t landed: if a newer tile (t+1) is still in flight its 8 async
    // loads may be outstanding (in-order completion), so wait <= 8; on the
    // last tile wait to zero.
    if (t < NTILES - 1)
      asm volatile("s_wait_asynccnt 0x8" ::: "memory");
    else
      asm volatile("s_wait_asynccnt 0x0" ::: "memory");
    __syncthreads();
    // Issue tile t+2 into buffer (t+2)&3; its previous readers (tile t-2)
    // retired before barrier t-1.
    if (t + 2 < NTILES) a_issue((t + 2) * KT, (t + 2) & (NBUF - 1));

    // A frag (16x32 bf16): chunks at K = 8*lhi and 16+8*lhi  (ISA 7.12.2)
    // B frag (32x16 bf16): chunks at K = 8*lhi and 16+8*lhi (same as A)
    const int a0 = 8 * lhi, a1 = 16 + 8 * lhi;
    const int b0 = 8 * lhi, b1 = 16 + 8 * lhi;
    v16bf fah[2], fal[2];
#pragma unroll
    for (int i = 0; i < 2; ++i) {
      const int r = 32 * wm + 16 * i + l15;
      fah[i] = frag2(&Ah[cur][r][a0], &Ah[cur][r][a1]);
      fal[i] = frag2(&Al[cur][r][a0], &Al[cur][r][a1]);
    }
#pragma unroll
    for (int j = 0; j < 4; ++j) {
      const int r = 64 * wn + 16 * j + l15;
      const v16bf fbh = frag2(&Bh[cur][r][b0], &Bh[cur][r][b1]);
      const v16bf fbl = frag2(&Bl[cur][r][b0], &Bl[cur][r][b1]);
      acc[0][j] = wmma32(fah[0], fbh, acc[0][j]);
      acc[1][j] = wmma32(fah[1], fbh, acc[1][j]);
      acc[0][j] = wmma32(fah[0], fbl, acc[0][j]);
      acc[1][j] = wmma32(fah[1], fbl, acc[1][j]);
      acc[0][j] = wmma32(fal[0], fbh, acc[0][j]);
      acc[1][j] = wmma32(fal[1], fbh, acc[1][j]);
    }
  }

#pragma unroll
  for (int j = 0; j < 4; ++j) {
    const int n = n0 + 64 * wn + 16 * j + l15;
    const float bv = bias[n];
#pragma unroll
    for (int i = 0; i < 2; ++i) {
      const int mb = m0 + 32 * wm + 16 * i + 8 * lhi;
#pragma unroll
      for (int r = 0; r < 8; ++r) {
        float v = acc[i][j][r] + bv;
        if (act) v = v / (1.f + expf(-v));
        out[(size_t)(mb + r) * N + n] = v;
      }
    }
  }
}

// ---------------------------------------------------------------------------
// Fallback exact-fp32 GEMM (V_WMMA_F32_16X16X4_F32) if ws is too small.
// ---------------------------------------------------------------------------
#define TK 16
#define LDSS (TK + 4)

__global__ __launch_bounds__(256)
void proj_gemm_f32(const float* __restrict__ X, const float* __restrict__ W,
                   const float* __restrict__ bias, float* __restrict__ out,
                   int act) {
  __shared__ float Afl[TM][LDSS];
  __shared__ float Bfl[TN][LDSS];

  const int K = D_DIM, N = D_DIM;
  const int tid = threadIdx.x;
  const int m0 = blockIdx.y * TM;
  const int n0 = blockIdx.x * TN;
  const int lr = tid >> 2;
  const int lc = (tid & 3) << 2;
  const int wave = tid >> 5;
  const int lane = tid & 31;
  const int wm = wave >> 1;
  const int wn = wave & 1;
  const int l15 = lane & 15;
  const int lhi = lane >> 4;

  v8f acc[2][4] = {};

  for (int k0 = 0; k0 < K; k0 += TK) {
#pragma unroll
    for (int i = 0; i < 2; ++i) {
      const int row = lr + 64 * i;
      const float4 va = *(const float4*)(X + (size_t)(m0 + row) * K + k0 + lc);
      *(float2*)&Afl[row][lc]     = make_float2(va.x, va.y);
      *(float2*)&Afl[row][lc + 2] = make_float2(va.z, va.w);
      const float4 vb = *(const float4*)(W + (size_t)(n0 + row) * K + k0 + lc);
      *(float2*)&Bfl[row][lc]     = make_float2(vb.x, vb.y);
      *(float2*)&Bfl[row][lc + 2] = make_float2(vb.z, vb.w);
    }
    __syncthreads();
#pragma unroll
    for (int kk = 0; kk < TK; kk += 4) {
      const int kp = kk + 2 * lhi;
      v2f a0 = *(const v2f*)&Afl[32 * wm + l15][kp];
      v2f a1 = *(const v2f*)&Afl[32 * wm + 16 + l15][kp];
      v2f b0 = *(const v2f*)&Bfl[64 * wn + l15][kp];
      v2f b1 = *(const v2f*)&Bfl[64 * wn + 16 + l15][kp];
      v2f b2 = *(const v2f*)&Bfl[64 * wn + 32 + l15][kp];
      v2f b3 = *(const v2f*)&Bfl[64 * wn + 48 + l15][kp];
      acc[0][0] = wmma4(a0, b0, acc[0][0]);
      acc[0][1] = wmma4(a0, b1, acc[0][1]);
      acc[0][2] = wmma4(a0, b2, acc[0][2]);
      acc[0][3] = wmma4(a0, b3, acc[0][3]);
      acc[1][0] = wmma4(a1, b0, acc[1][0]);
      acc[1][1] = wmma4(a1, b1, acc[1][1]);
      acc[1][2] = wmma4(a1, b2, acc[1][2]);
      acc[1][3] = wmma4(a1, b3, acc[1][3]);
    }
    __syncthreads();
  }

#pragma unroll
  for (int j = 0; j < 4; ++j) {
    const int n = n0 + 64 * wn + 16 * j + l15;
    const float bv = bias[n];
#pragma unroll
    for (int i = 0; i < 2; ++i) {
      const int mb = m0 + 32 * wm + 16 * i + 8 * lhi;
#pragma unroll
      for (int r = 0; r < 8; ++r) {
        float v = acc[i][j][r] + bv;
        if (act) v = v / (1.f + expf(-v));
        out[(size_t)(mb + r) * N + n] = v;
      }
    }
  }
}

// ---------------------------------------------------------------------------
// Diagonal recurrence with a depth-8 software pipeline (only ~1 wave/SIMD is
// resident, so latency must be hidden in-wave).  Aliasing is safe by
// construction: loads at t+8 (h-slot t+9 / out[t+8]) never touch stores <= t.
// ---------------------------------------------------------------------------
#define PIPE 8

__global__ __launch_bounds__(256)
void scan_fused(const float* __restrict__ h0, const float* __restrict__ A_log,
                const float* __restrict__ r_delta,
                const float* xv, const float* xd, const float* gate,
                float* out, float* hout) {
  const int j = blockIdx.x * blockDim.x + threadIdx.x;
  const int d = j & (D_DIM - 1);
  const float rh = -expf(A_log[d]);
  const float rd = r_delta[d];
  float h = h0[j];
  __builtin_nontemporal_store(h, &hout[j]);
  const int S = B_DIM * D_DIM;

  float pa[PIPE], pd[PIPE], pg[PIPE];
#pragma unroll
  for (int p = 0; p < PIPE; ++p) {
    const size_t idx = (size_t)p * S + j;
    pa[p] = xv[idx];
    pd[p] = xd[idx];
    pg[p] = gate[idx];
  }

  for (int t0 = 0; t0 < T_DIM; t0 += PIPE) {
#pragma unroll
    for (int p = 0; p < PIPE; ++p) {
      const int t = t0 + p;
      const float a = pa[p], dl = pd[p], g = pg[p];
      if (t + PIPE < T_DIM) {
        const size_t nidx = (size_t)(t + PIPE) * S + j;
        pa[p] = xv[nidx];
        pd[p] = xd[nidx];
        pg[p] = gate[nidx];
      }
      const float v = rh * h + a;
      const float delta = 1.f / (1.f + expf(-(dl + rd * h)));
      h = (1.f - delta) * h + delta * tanhf(v);
      const size_t idx = (size_t)t * S + j;
      __builtin_nontemporal_store(h * g, &out[idx]);
      __builtin_nontemporal_store(h, &hout[idx + S]);
    }
  }
}

// ---------------------------------------------------------------------------
extern "C" void kernel_launch(void* const* d_in, const int* in_sizes, int n_in,
                              void* d_out, int out_size, void* d_ws,
                              size_t ws_size, hipStream_t stream) {
  const float* x       = (const float*)d_in[0];
  const float* h0      = (const float*)d_in[1];
  const float* A_log   = (const float*)d_in[2];
  const float* r_delta = (const float*)d_in[3];
  const float* W_x     = (const float*)d_in[4];
  const float* W_delta = (const float*)d_in[5];
  const float* W_gate  = (const float*)d_in[6];
  const float* b       = (const float*)d_in[7];
  const float* b_delta = (const float*)d_in[8];
  const float* b_gate  = (const float*)d_in[9];

  const size_t TBD = (size_t)T_DIM * B_DIM * D_DIM;
  const size_t BD  = (size_t)B_DIM * D_DIM;
  const size_t DD  = (size_t)D_DIM * D_DIM;

  float* out      = (float*)d_out;   // [T,B,D]
  float* hout     = out + TBD;       // [T+1,B,D]
  float* xv_buf   = hout + BD;       // aliases h[1..T]
  float* gate_buf = out;             // aliases output
  float* xd_buf   = (float*)d_ws;    // T*B*D floats

  const dim3 grid(D_DIM / TN, M_DIM / TM);  // (8, 256)

  const size_t need = TBD * 4 + TBD * 4 + 3 * DD * 4;
  if (ws_size >= need) {
    unsigned short* xh  = (unsigned short*)(xd_buf + TBD);
    unsigned short* xl  = xh + TBD;
    unsigned short* wxh = xl + TBD;
    unsigned short* wxl = wxh + DD;
    unsigned short* wdh = wxl + DD;
    unsigned short* wdl = wdh + DD;
    unsigned short* wgh = wdl + DD;
    unsigned short* wgl = wgh + DD;

    const int nx4 = (int)(TBD / 4);
    const int nw4 = (int)(DD / 4);
    split_kernel<<<(nx4 + 255) / 256, 256, 0, stream>>>(x, xh, xl, nx4);
    split_kernel<<<(nw4 + 255) / 256, 256, 0, stream>>>(W_x, wxh, wxl, nw4);
    split_kernel<<<(nw4 + 255) / 256, 256, 0, stream>>>(W_delta, wdh, wdl, nw4);
    split_kernel<<<(nw4 + 255) / 256, 256, 0, stream>>>(W_gate, wgh, wgl, nw4);

    proj_gemm_bf3<<<grid, 256, 0, stream>>>(xh, xl, wxh, wxl, b, xv_buf, 0);
    proj_gemm_bf3<<<grid, 256, 0, stream>>>(xh, xl, wdh, wdl, b_delta, xd_buf, 0);
    proj_gemm_bf3<<<grid, 256, 0, stream>>>(xh, xl, wgh, wgl, b_gate, gate_buf, 1);
  } else {
    proj_gemm_f32<<<grid, 256, 0, stream>>>(x, W_x, b, xv_buf, 0);
    proj_gemm_f32<<<grid, 256, 0, stream>>>(x, W_delta, b_delta, xd_buf, 0);
    proj_gemm_f32<<<grid, 256, 0, stream>>>(x, W_gate, b_gate, gate_buf, 1);
  }

  scan_fused<<<(B_DIM * D_DIM) / 256, 256, 0, stream>>>(
      h0, A_log, r_delta, xv_buf, xd_buf, gate_buf, out, hout);
}
